// MultiHeadAttention_26774826123563
// MI455X (gfx1250) — matrix operand order repaired and checked by reading, not compiled, not run on hardware
//
#include <hip/hip_runtime.h>

typedef __attribute__((ext_vector_type(16))) __bf16       v16bf;
typedef __attribute__((ext_vector_type(8)))  float        v8f;
typedef __attribute__((ext_vector_type(4)))  unsigned int u32x4;

union Frag16 { u32x4 q[2]; v16bf h; };

__device__ __forceinline__ unsigned short f2bf(float f) {
  union { float f; unsigned u; } v; v.f = f;
  unsigned r = v.u + 0x7FFFu + ((v.u >> 16) & 1u);   // round-to-nearest-even
  return (unsigned short)(r >> 16);
}

__device__ __forceinline__ v8f wmma_bf16(const Frag16& a, const Frag16& b, v8f c) {
  return __builtin_amdgcn_wmma_f32_16x16x32_bf16(false, a.h, false, b.h,
                                                 (short)0, c, false, false);
}

// ---- CDNA5 async global->LDS (GV mode, per-lane 16B), tracked by ASYNCcnt ----
__device__ __forceinline__ void async_ld_b128(void* lds_ptr, const void* gptr) {
  unsigned loff = (unsigned)(uintptr_t)lds_ptr;   // low 32 bits = LDS offset
  asm volatile("global_load_async_to_lds_b128 %0, %1, off"
               :: "v"(loff), "v"(gptr) : "memory");
}
__device__ __forceinline__ void wait_async_le4() {
  asm volatile("s_wait_asynccnt 0x4" ::: "memory");
}
__device__ __forceinline__ void wait_async_0() {
  asm volatile("s_wait_asynccnt 0x0" ::: "memory");
}

// ---------------- elementwise f32 -> bf16 ----------------
__global__ void cvt_bf16_kernel(const float* __restrict__ in,
                                unsigned short* __restrict__ out, int n) {
  int i = blockIdx.x * blockDim.x + threadIdx.x;
  if (i < n) out[i] = f2bf(in[i]);
}

// -------- transpose + convert: out[b][c][r] = bf16(in[b][r][c]) --------
__global__ void transpose_bf16_kernel(const float* __restrict__ in,
                                      unsigned short* __restrict__ out,
                                      int R, int C) {
  long bofs = (long)blockIdx.y * R * C;
  long total = (long)R * C;
  for (long i = blockIdx.x * (long)blockDim.x + threadIdx.x; i < total;
       i += (long)gridDim.x * blockDim.x) {
    int c = (int)(i / R), r = (int)(i % R);
    out[bofs + i] = f2bf(in[bofs + (long)r * C + c]);
  }
}

// ---------------- WMMA GEMM: C = A(bf16,row-major) x Bt(bf16,N-major)^T + bias ----
#define BM 128
#define BN 128
#define BK 32
#define LPAD 40   // padded LDS row stride (halfs): 80B = 20 dwords -> conflict-free

__global__ __launch_bounds__(256) void gemm_bf16_kernel(
    const unsigned short* __restrict__ A,  int lda, long sA,
    const unsigned short* __restrict__ Bt, int ldb, long sB,
    const float* __restrict__ bias, int sBias,
    void* __restrict__ C, int ldc, long sC,
    int K, int out_bf16)
{
  __shared__ unsigned short Al[2][BM * LPAD];
  __shared__ unsigned short Bl[2][BN * LPAD];

  const int z = blockIdx.z;
  const unsigned short* Ag = A  + z * sA + (long)blockIdx.x * BM * lda;
  const unsigned short* Bg = Bt + z * sB + (long)blockIdx.y * BN * ldb;

  const int tid = threadIdx.x;
  const int wid = tid >> 5, l32 = tid & 31, hh = l32 >> 4, ln = l32 & 15;
  const int wm = wid & 3, wn = wid >> 2;   // 4(M) x 2(N) waves, each 32x64

  // this thread's two 16B chunks of each 128x32 tile
  const int rA0 = tid >> 2,          kA0 = (tid & 3) << 3;
  const int rA1 = (tid + 256) >> 2,  kA1 = ((tid + 256) & 3) << 3;

  const v8f vzero = {0.f,0.f,0.f,0.f,0.f,0.f,0.f,0.f};
  v8f acc[2][4];
  for (int i = 0; i < 2; ++i) for (int j = 0; j < 4; ++j) acc[i][j] = vzero;

  // prologue: async-fill buffer 0 (4 async ops per wave)
  {
    int k0 = 0;
    async_ld_b128(&Al[0][rA0 * LPAD + kA0], &Ag[(long)rA0 * lda + k0 + kA0]);
    async_ld_b128(&Al[0][rA1 * LPAD + kA1], &Ag[(long)rA1 * lda + k0 + kA1]);
    async_ld_b128(&Bl[0][rA0 * LPAD + kA0], &Bg[(long)rA0 * ldb + k0 + kA0]);
    async_ld_b128(&Bl[0][rA1 * LPAD + kA1], &Bg[(long)rA1 * ldb + k0 + kA1]);
  }

  int buf = 0;
  for (int k0 = 0; k0 < K; k0 += BK) {
    __syncthreads();          // everyone done reading buf^1 -> safe to overwrite
    if (k0 + BK < K) {
      int kn = k0 + BK, nb = buf ^ 1;
      async_ld_b128(&Al[nb][rA0 * LPAD + kA0], &Ag[(long)rA0 * lda + kn + kA0]);
      async_ld_b128(&Al[nb][rA1 * LPAD + kA1], &Ag[(long)rA1 * lda + kn + kA1]);
      async_ld_b128(&Bl[nb][rA0 * LPAD + kA0], &Bg[(long)rA0 * ldb + kn + kA0]);
      async_ld_b128(&Bl[nb][rA1 * LPAD + kA1], &Bg[(long)rA1 * ldb + kn + kA1]);
      wait_async_le4();       // in-order completion: current tile's 4 ops done
    } else {
      wait_async_0();
    }
    __syncthreads();          // all waves' async data landed in LDS

    Frag16 af[2], bf[4];
    for (int i = 0; i < 2; ++i) {
      int row = wm * 32 + i * 16 + ln;                 // A: lane = row m
      af[i].q[0] = *(const u32x4*)&Al[buf][row * LPAD + hh * 8];
      af[i].q[1] = *(const u32x4*)&Al[buf][row * LPAD + 16 + hh * 8];
    }
    for (int j = 0; j < 4; ++j) {
      int n = wn * 64 + j * 16 + ln;                   // B: lane = column n
      bf[j].q[0] = *(const u32x4*)&Bl[buf][n * LPAD + hh * 8];
      bf[j].q[1] = *(const u32x4*)&Bl[buf][n * LPAD + 16 + hh * 8];
    }
    for (int i = 0; i < 2; ++i)
      for (int j = 0; j < 4; ++j)
        acc[i][j] = wmma_bf16(af[i], bf[j], acc[i][j]);
    buf ^= 1;
  }

  for (int j = 0; j < 4; ++j) {
    int colg = blockIdx.y * BN + wn * 64 + j * 16 + ln;
    float bv = bias ? bias[(long)z * sBias + colg] : 0.f;
    for (int i = 0; i < 2; ++i)
      for (int r = 0; r < 8; ++r) {
        int rowg = blockIdx.x * BM + wm * 32 + i * 16 + r + hh * 8;
        float val = acc[i][j][r] + bv;
        long idx = z * sC + (long)rowg * ldc + colg;
        if (out_bf16) ((unsigned short*)C)[idx] = f2bf(val);
        else          ((float*)C)[idx] = val;
      }
  }
}

// ---------------- flash attention per (s-tile, b, h) ----------------
// q,k,v: bf16 [H][B][S][D]; heads out: bf16 [B*S][H*D]
__global__ __launch_bounds__(256) void attn_kernel(
    const unsigned short* __restrict__ q,
    const unsigned short* __restrict__ k,
    const unsigned short* __restrict__ v,
    unsigned short* __restrict__ heads)
{
  const int S = 1024, D = 512, Bz = 8;
  extern __shared__ unsigned short sm[];
  unsigned short* qs  = sm;                              // 64 x 520
  unsigned short* ks  = sm + 64 * 520;                   // 64 x 520
  unsigned short* vs  = sm + 2 * 64 * 520;               // vT: 512 x 72
  unsigned short* psm = sm + 2 * 64 * 520 + 512 * 72;    // P: 4 x 16 x 72

  const int st = blockIdx.x, b = blockIdx.y, h = blockIdx.z;
  const int tid = threadIdx.x;
  const int wid = tid >> 5, l32 = tid & 31, hh = l32 >> 4, ln = l32 & 15;
  const int rg = wid >> 1, dh = wid & 1;  // 4 row-groups x 2 D-halves

  const unsigned short* qg = q + (((long)(h * Bz + b)) * S + st * 64) * D;
  const unsigned short* kg = k + ((long)(h * Bz + b)) * S * D;
  const unsigned short* vg = v + ((long)(h * Bz + b)) * S * D;

  // resident q tile: 64 x 512 halfs, async into LDS (awaited at first barrier)
  for (int c = tid; c < 4096; c += 256) {
    int row = c >> 6, kc = (c & 63) << 3;
    async_ld_b128(&qs[row * 520 + kc], &qg[(long)row * D + kc]);
  }

  const v8f vzero = {0.f,0.f,0.f,0.f,0.f,0.f,0.f,0.f};
  v8f oacc[16];
  for (int j = 0; j < 16; ++j) oacc[j] = vzero;
  float row_max[8], row_sum[8];
  for (int r = 0; r < 8; ++r) { row_max[r] = -INFINITY; row_sum[r] = 0.f; }
  const float scale = 0.044194173824159216f;  // 1/sqrt(512)

  for (int t0 = 0; t0 < S; t0 += 64) {
    __syncthreads();
    for (int c = tid; c < 4096; c += 256) {        // K tile: async to LDS
      int row = c >> 6, kc = (c & 63) << 3;
      async_ld_b128(&ks[row * 520 + kc], &kg[((long)(t0 + row)) * D + kc]);
    }
    for (int c = tid; c < 4096; c += 256) {        // V tile: transpose via VGPRs
      int t = c >> 6, dc = (c & 63) << 3;
      union { u32x4 v4; unsigned short s[8]; } tu;
      tu.v4 = *(const u32x4*)&vg[((long)(t0 + t)) * D + dc];
      for (int e = 0; e < 8; ++e) vs[(dc + e) * 72 + t] = tu.s[e];
    }
    wait_async_0();
    __syncthreads();

    // S = q . k^T  (16 rows x 64 cols per wave, K=512)
    v8f sacc[4];
    for (int j = 0; j < 4; ++j) sacc[j] = vzero;
    const int qrow = rg * 16 + ln;
    for (int kk = 0; kk < D; kk += 32) {
      Frag16 a;
      a.q[0] = *(const u32x4*)&qs[qrow * 520 + kk + hh * 8];
      a.q[1] = *(const u32x4*)&qs[qrow * 520 + kk + 16 + hh * 8];
      for (int j = 0; j < 4; ++j) {
        Frag16 bfr;
        bfr.q[0] = *(const u32x4*)&ks[(j * 16 + ln) * 520 + kk + hh * 8];
        bfr.q[1] = *(const u32x4*)&ks[(j * 16 + ln) * 520 + kk + 16 + hh * 8];
        sacc[j] = wmma_bf16(a, bfr, sacc[j]);
      }
    }
    for (int j = 0; j < 4; ++j) sacc[j] *= scale;

    // online softmax; row m = (r + 8*hh), columns span the 16 lanes of a half
    for (int r = 0; r < 8; ++r) {
      float m = fmaxf(fmaxf(sacc[0][r], sacc[1][r]), fmaxf(sacc[2][r], sacc[3][r]));
      for (int off = 8; off; off >>= 1) m = fmaxf(m, __shfl_xor(m, off, 32));
      float om = row_max[r];
      float nm = fmaxf(om, m);
      float alpha = __expf(om - nm);
      row_max[r] = nm;
      float pj[4], psum = 0.f;
      for (int j = 0; j < 4; ++j) { pj[j] = __expf(sacc[j][r] - nm); psum += pj[j]; }
      for (int off = 8; off; off >>= 1) psum += __shfl_xor(psum, off, 32);
      row_sum[r] = row_sum[r] * alpha + psum;
      for (int j2 = 0; j2 < 16; ++j2) oacc[j2][r] *= alpha;
      if (dh == 0) {   // one wave of the pair writes P (C-layout -> LDS row-major)
        int pbase = rg * 1152 + (r + 8 * hh) * 72;
        for (int j = 0; j < 4; ++j) psm[pbase + j * 16 + ln] = f2bf(pj[j]);
      }
    }
    __syncthreads();

    // O += P . V   (16 x 256 per wave, K=64 in two wmma steps)
    for (int kk2 = 0; kk2 < 64; kk2 += 32) {
      Frag16 a;
      a.q[0] = *(const u32x4*)&psm[rg * 1152 + ln * 72 + kk2 + hh * 8];
      a.q[1] = *(const u32x4*)&psm[rg * 1152 + ln * 72 + kk2 + 16 + hh * 8];
      for (int j2 = 0; j2 < 16; ++j2) {
        int dcol = dh * 256 + j2 * 16 + ln;
        Frag16 bfr;
        bfr.q[0] = *(const u32x4*)&vs[dcol * 72 + kk2 + hh * 8];
        bfr.q[1] = *(const u32x4*)&vs[dcol * 72 + kk2 + 16 + hh * 8];
        oacc[j2] = wmma_bf16(a, bfr, oacc[j2]);
      }
    }
  }

  // epilogue: normalize and scatter into concat layout [B*S][H*D]
  for (int j2 = 0; j2 < 16; ++j2)
    for (int r = 0; r < 8; ++r) {
      float val = oacc[j2][r] / row_sum[r];
      int srow = st * 64 + rg * 16 + r + 8 * hh;
      int f = h * 512 + dh * 256 + j2 * 16 + ln;
      heads[((long)(b * S + srow)) * 4096 + f] = f2bf(val);
    }
}

extern "C" void kernel_launch(void* const* d_in, const int* in_sizes, int n_in,
                              void* d_out, int out_size, void* d_ws, size_t ws_size,
                              hipStream_t stream) {
  (void)in_sizes; (void)n_in; (void)out_size; (void)ws_size;
  const float* x  = (const float*)d_in[0];
  const float* Wq = (const float*)d_in[1];
  const float* bq = (const float*)d_in[2];
  const float* Wk = (const float*)d_in[3];
  const float* bk = (const float*)d_in[4];
  const float* Wv = (const float*)d_in[5];
  const float* bv = (const float*)d_in[6];
  const float* Wo = (const float*)d_in[7];
  const float* bo = (const float*)d_in[8];
  float* out = (float*)d_out;

  const int Bz = 8, S = 1024, D = 512, H = 8;
  const long M  = (long)Bz * S;   // 8192
  const long HD = (long)H * D;    // 4096

  char* ws = (char*)d_ws;
  size_t o = 0;
  auto alloc = [&](size_t bytes) {
    char* p = ws + o; o += (bytes + 255) & ~(size_t)255; return p;
  };
  unsigned short* xbf = (unsigned short*)alloc((size_t)M * D * 2);
  unsigned short* wqT = (unsigned short*)alloc((size_t)H * D * D * 2);
  unsigned short* wkT = (unsigned short*)alloc((size_t)H * D * D * 2);
  unsigned short* wvT = (unsigned short*)alloc((size_t)H * D * D * 2);
  unsigned short* woT = (unsigned short*)alloc((size_t)HD * D * 2);
  unsigned short* qws = (unsigned short*)alloc((size_t)H * M * D * 2);
  unsigned short* kws = (unsigned short*)alloc((size_t)H * M * D * 2);
  unsigned short* vws = (unsigned short*)alloc((size_t)H * M * D * 2);
  unsigned short* hws = (unsigned short*)alloc((size_t)M * HD * 2);

  {
    int n = (int)(M * D);
    cvt_bf16_kernel<<<dim3((n + 255) / 256), dim3(256), 0, stream>>>(x, xbf, n);
  }
  transpose_bf16_kernel<<<dim3(1024, H), dim3(256), 0, stream>>>(Wq, wqT, D, D);
  transpose_bf16_kernel<<<dim3(1024, H), dim3(256), 0, stream>>>(Wk, wkT, D, D);
  transpose_bf16_kernel<<<dim3(1024, H), dim3(256), 0, stream>>>(Wv, wvT, D, D);
  transpose_bf16_kernel<<<dim3(8192, 1), dim3(256), 0, stream>>>(Wo, woT, (int)HD, D);

  dim3 gq((unsigned)(M / 128), (unsigned)(D / 128), (unsigned)H);
  gemm_bf16_kernel<<<gq, 256, 0, stream>>>(xbf, D, 0L, wqT, D, (long)D * D, bq, D,
                                           qws, D, M * D, D, 1);
  gemm_bf16_kernel<<<gq, 256, 0, stream>>>(xbf, D, 0L, wkT, D, (long)D * D, bk, D,
                                           kws, D, M * D, D, 1);
  gemm_bf16_kernel<<<gq, 256, 0, stream>>>(xbf, D, 0L, wvT, D, (long)D * D, bv, D,
                                           vws, D, M * D, D, 1);

  size_t smem = (size_t)(2 * 64 * 520 + 512 * 72 + 4 * 16 * 72) * 2;  // ~216 KB
  attn_kernel<<<dim3(S / 64, Bz, H), dim3(256), smem, stream>>>(qws, kws, vws, hws);

  dim3 go((unsigned)(M / 128), (unsigned)(D / 128), 1);
  gemm_bf16_kernel<<<go, 256, 0, stream>>>(hws, (int)HD, 0L, woT, (int)HD, 0L, bo, 0,
                                           out, D, 0L, (int)HD, 0);
}
